// MoE_10943576670937
// MI455X (gfx1250) — matrix operand order repaired and checked by reading, not compiled, not run on hardware
//
#include <hip/hip_runtime.h>

#define N_TOK 4096
#define DIM   1024
#define DFF   4096
#define NE    8

typedef _Float16 v16h __attribute__((ext_vector_type(16)));
typedef _Float16 v8h  __attribute__((ext_vector_type(8)));
typedef float    v8f  __attribute__((ext_vector_type(8)));
typedef float    v4f  __attribute__((ext_vector_type(4)));

__device__ __forceinline__ v8f wmma_f16(v16h a, v16h b, v8f c) {
  // (neg_a, A, neg_b, B, c_mod, C, reuse_a, reuse_b)
  return __builtin_amdgcn_wmma_f32_16x16x32_f16(false, a, false, b, (short)0, c, false, false);
}

// async global->LDS copy of 16B per lane (CDNA5, ASYNCcnt-tracked)
#define ASYNC_B128(ldsoff, gptr)                                              \
  asm volatile("global_load_async_to_lds_b128 %0, %1, off"                    \
               :: "v"(ldsoff), "v"(gptr) : "memory")
#define WAIT_ASYNC_LE4()  asm volatile("s_wait_asynccnt 0x4" ::: "memory")
#define WAIT_ASYNC_0()    asm volatile("s_wait_asynccnt 0x0" ::: "memory")
#define WAIT_DS_0()       asm volatile("s_wait_dscnt 0x0"    ::: "memory")

// ---------------- kernel 0: zero per-expert counters ----------------
__global__ void moe_zero_counts(int* __restrict__ counts) {
  if (threadIdx.x < NE) counts[threadIdx.x] = 0;
}

// ---------------- kernel 1: gate logits + argmax + scatter ----------------
__global__ __launch_bounds__(256) void moe_gate(
    const float* __restrict__ x, const float* __restrict__ gw,
    const float* __restrict__ gb, int* __restrict__ counts,
    int* __restrict__ list) {
  __shared__ float sgw[DIM * NE];  // 32 KB
  int tid = threadIdx.x;
  for (int i = tid * 4; i < DIM * NE; i += 256 * 4)
    *(v4f*)(sgw + i) = *(const v4f*)(gw + i);
  __syncthreads();

  int tok = blockIdx.x * 256 + tid;
  float acc[NE];
#pragma unroll
  for (int e = 0; e < NE; ++e) acc[e] = gb[e];
  const float* xr = x + (size_t)tok * DIM;
  for (int k = 0; k < DIM; k += 4) {
    v4f xq = *(const v4f*)(xr + k);
#pragma unroll
    for (int j = 0; j < 4; ++j) {
#pragma unroll
      for (int e = 0; e < NE; ++e) acc[e] += xq[j] * sgw[(k + j) * NE + e];
    }
  }
  int best = 0; float bv = acc[0];
#pragma unroll
  for (int e = 1; e < NE; ++e)
    if (acc[e] > bv) { bv = acc[e]; best = e; }  // strict > == first-max argmax
  int slot = atomicAdd(&counts[best], 1);
  list[best * N_TOK + slot] = tok;
}

// ---------------- kernel 2: grouped expert FFN via f16 WMMA ----------------
#define LXS (DIM + 8)   // padded halfword stride for X tile (conflict-free, 16B aligned)
#define LHS (DFF + 8)   // padded halfword stride for H tile
#define STS 20          // stage row stride in floats (80B: 16B-aligned b128, conflict-free)

__global__ __launch_bounds__(256) void moe_expert(
    const float* __restrict__ x,  const float* __restrict__ w1,
    const float* __restrict__ b1, const float* __restrict__ w2,
    const float* __restrict__ b2, float* __restrict__ out,
    const int* __restrict__ counts, const int* __restrict__ list) {
  __shared__ _Float16 lX[16 * LXS];         // 33 KB  X tile, f16, [token][k]
  __shared__ _Float16 lH[16 * LHS];         // 128 KB H tile, f16, [token][ff]
  __shared__ float    lS[8][2][32 * STS];   // 40 KB  per-wave double-buffered fp32 stage

  int e    = blockIdx.x >> 8;
  int tile = blockIdx.x & 255;
  int cnt  = counts[e];
  int base = tile * 16;
  if (base >= cnt) return;                  // uniform early exit for empty tiles
  int vn = cnt - base; if (vn > 16) vn = 16;

  int tid  = threadIdx.x;
  int lane = tid & 31;
  int wave = tid >> 5;
  const int* tl = list + e * N_TOK + base;

  // ---- load X tile (16 tokens x 1024) fp32 -> f16 into LDS ----
  {
    int trow = tid >> 4;                    // 0..15 token row in tile
    int cc   = (tid & 15) * 64;             // 64-float chunk
    int idx  = trow < vn ? trow : vn - 1;   // clamp padding rows (keeps EXEC full)
    int gtok = tl[idx];
    const float* xr = x + (size_t)gtok * DIM + cc;
    _Float16* dst = lX + trow * LXS + cc;
#pragma unroll
    for (int i = 0; i < 64; i += 8) {
      v4f a = *(const v4f*)(xr + i);
      v4f b = *(const v4f*)(xr + i + 4);
      v8h h;
#pragma unroll
      for (int j = 0; j < 4; ++j) { h[j] = (_Float16)a[j]; h[j + 4] = (_Float16)b[j]; }
      *(v8h*)(dst + i) = h;
    }
  }
  __syncthreads();

  int half = lane >> 4;     // 0: lanes 0-15, 1: lanes 16-31
  int col  = lane & 15;     // A: M index / B: N index / D: N index
  int klo  = half * 16;     // A-operand K sub-offset (contiguous, matches B)
  int hoff = half * 8;      // D-operand M offset per ISA layout

  // async-stage lane geometry: each of 4 issues copies rows i*8+ln4, cols ccq..+3
  int ln4 = lane >> 2;                      // 0..7
  int ccq = (lane & 3) * 4;                 // 0,4,8,12
  const uint32_t BUFB = 32 * STS * 4;       // bytes per stage buffer
  const uint32_t ROWB = 8 * STS * 4;        // bytes per 8 stage rows
  // flat shared ptr low 32 bits == LDS byte offset (LDS aperture: addr[31:0])
  uint32_t lbase   = (uint32_t)(uintptr_t)(&lS[wave][0][0]);
  uint32_t laneoff = (uint32_t)((ln4 * STS + ccq) * 4);

  // ======== Phase A: H = relu(X @ W1 + b1), D oriented [M=ff][N=token] ========
  const float* w1e = w1 + (size_t)e * DIM * DFF;
  {
    const size_t rowoff = (size_t)ln4 * DFF + ccq;
    // prologue: prefetch chunk (mt=wave, k0=0) -> buffer 0
    WAIT_DS_0();
    {
      const float* s0 = w1e + (size_t)wave * 16 + rowoff;
      uint32_t lo = lbase + laneoff;
      ASYNC_B128(lo,            s0);
      ASYNC_B128(lo + ROWB,     s0 +  8 * (size_t)DFF);
      ASYNC_B128(lo + 2 * ROWB, s0 + 16 * (size_t)DFF);
      ASYNC_B128(lo + 3 * ROWB, s0 + 24 * (size_t)DFF);
    }
    int pb = 0;
    for (int mt = wave; mt < DFF / 16; mt += 8) {
      int m0 = mt * 16;
      v8f c;
      const float* bb = b1 + (size_t)e * DFF + m0 + hoff;
#pragma unroll
      for (int r = 0; r < 8; ++r) c[r] = bb[r];   // accumulate on top of bias
      for (int k0 = 0; k0 < DIM; k0 += 32) {
        // next chunk coords (last iteration re-issues current chunk: harmless)
        int nmt = mt, nk = k0 + 32;
        if (nk == DIM) { nk = 0; nmt = mt + 8; if (nmt >= DFF / 16) { nmt = mt; nk = k0; } }
        WAIT_DS_0();                                // WAR: prior gathers of buf pb^1 done
        {
          const float* s0 = w1e + (size_t)nk * DFF + (size_t)nmt * 16 + rowoff;
          uint32_t lo = lbase + (uint32_t)(pb ^ 1) * BUFB + laneoff;
          ASYNC_B128(lo,            s0);
          ASYNC_B128(lo + ROWB,     s0 +  8 * (size_t)DFF);
          ASYNC_B128(lo + 2 * ROWB, s0 + 16 * (size_t)DFF);
          ASYNC_B128(lo + 3 * ROWB, s0 + 24 * (size_t)DFF);
        }
        WAIT_ASYNC_LE4();                           // in-order: previous 4 have landed
        const float* st = &lS[wave][pb][0];
        v16h a;                                     // A = W1^T, contiguous k order (matches B)
#pragma unroll
        for (int j = 0; j < 8; ++j) {
          a[j]     = (_Float16)st[(klo + j) * STS + col];
          a[j + 8] = (_Float16)st[(klo + 8 + j) * STS + col];
        }
        const v8h* xb = (const v8h*)(lX + col * LXS + k0 + half * 16);  // B = X^T
        v8h blo = xb[0], bhi = xb[1];
        v16h b;
#pragma unroll
        for (int i = 0; i < 8; ++i) { b[i] = blo[i]; b[i + 8] = bhi[i]; }
        c = wmma_f16(a, b, c);
        pb ^= 1;
      }
      v8h hh;                                       // relu + pack -> one ds_store_b128
#pragma unroll
      for (int r = 0; r < 8; ++r) { float v = c[r]; hh[r] = (_Float16)(v > 0.f ? v : 0.f); }
      *(v8h*)(lH + col * LHS + m0 + hoff) = hh;     // H[token=col][ff m0+hoff..+7]
    }
    WAIT_ASYNC_0();                                 // drain junk prefetch before reuse
  }
  __syncthreads();

  // ======== Phase B: Y = H @ W2 + b2, D oriented [M=outcol][N=token] ========
  const float* w2e = w2 + (size_t)e * DFF * DIM;
  {
    const size_t rowoff = (size_t)ln4 * DIM + ccq;
    WAIT_DS_0();
    {
      const float* s0 = w2e + (size_t)wave * 16 + rowoff;
      uint32_t lo = lbase + laneoff;
      ASYNC_B128(lo,            s0);
      ASYNC_B128(lo + ROWB,     s0 +  8 * (size_t)DIM);
      ASYNC_B128(lo + 2 * ROWB, s0 + 16 * (size_t)DIM);
      ASYNC_B128(lo + 3 * ROWB, s0 + 24 * (size_t)DIM);
    }
    int pb = 0;
    for (int mt = wave; mt < DIM / 16; mt += 8) {
      int m0 = mt * 16;
      v8f c;
      const float* bb = b2 + (size_t)e * DIM + m0 + hoff;
#pragma unroll
      for (int r = 0; r < 8; ++r) c[r] = bb[r];
      for (int k0 = 0; k0 < DFF; k0 += 32) {
        int nmt = mt, nk = k0 + 32;
        if (nk == DFF) { nk = 0; nmt = mt + 8; if (nmt >= DIM / 16) { nmt = mt; nk = k0; } }
        WAIT_DS_0();
        {
          const float* s0 = w2e + (size_t)nk * DIM + (size_t)nmt * 16 + rowoff;
          uint32_t lo = lbase + (uint32_t)(pb ^ 1) * BUFB + laneoff;
          ASYNC_B128(lo,            s0);
          ASYNC_B128(lo + ROWB,     s0 +  8 * (size_t)DIM);
          ASYNC_B128(lo + 2 * ROWB, s0 + 16 * (size_t)DIM);
          ASYNC_B128(lo + 3 * ROWB, s0 + 24 * (size_t)DIM);
        }
        WAIT_ASYNC_LE4();
        const float* st = &lS[wave][pb][0];
        v16h a;                                     // A = W2^T
#pragma unroll
        for (int j = 0; j < 8; ++j) {
          a[j]     = (_Float16)st[(klo + j) * STS + col];
          a[j + 8] = (_Float16)st[(klo + 8 + j) * STS + col];
        }
        const v8h* hb = (const v8h*)(lH + col * LHS + k0 + half * 16);  // B = H^T
        v8h blo = hb[0], bhi = hb[1];
        v16h b;
#pragma unroll
        for (int i = 0; i < 8; ++i) { b[i] = blo[i]; b[i + 8] = bhi[i]; }
        c = wmma_f16(a, b, c);
        pb ^= 1;
      }
      if (col < vn) {                               // divergence only after all WMMAs
        int gtok = tl[col];
        float* op = out + (size_t)gtok * DIM + m0 + hoff;
        v4f lo4, hi4;
#pragma unroll
        for (int r = 0; r < 4; ++r) { lo4[r] = c[r]; hi4[r] = c[r + 4]; }
        *(v4f*)op = lo4;
        *(v4f*)(op + 4) = hi4;
      }
    }
    // trailing junk prefetch drains at s_endpgm (implicit wait-idle)
  }
}

// ---------------- launch ----------------
extern "C" void kernel_launch(void* const* d_in, const int* in_sizes, int n_in,
                              void* d_out, int out_size, void* d_ws, size_t ws_size,
                              hipStream_t stream) {
  const float* x  = (const float*)d_in[0];
  const float* gw = (const float*)d_in[1];
  const float* gb = (const float*)d_in[2];
  const float* w1 = (const float*)d_in[3];
  const float* b1 = (const float*)d_in[4];
  const float* w2 = (const float*)d_in[5];
  const float* b2 = (const float*)d_in[6];
  float* out = (float*)d_out;

  int* counts = (int*)d_ws;        // 8 ints (zeroed every call)
  int* list   = counts + 16;       // 8 * 4096 token ids

  moe_zero_counts<<<1, 32, 0, stream>>>(counts);
  moe_gate<<<N_TOK / 256, 256, 0, stream>>>(x, gw, gb, counts, list);
  moe_expert<<<NE * 256, 256, 0, stream>>>(x, w1, b1, w2, b2, out, counts, list);
}
